// UnifiedEmbedding_4501125726693
// MI455X (gfx1250) — matrix operand order repaired and checked by reading, not compiled, not run on hardware
//
#include <hip/hip_runtime.h>
#include <hip/hip_bf16.h>

// ---------------------------------------------------------------------------
// UnifiedEmbedding forward for MI455X (gfx1250, wave32).
//   x   = tok_embed[idxs]            [16384, 2048]
//   h   = gelu(x @ W1^T + b1)        [16384, 1024]   (bf16 in workspace)
//   out = h @ W2^T + b2              [16384, 1024]   fp32
// Data movement per GEMM K-step (double buffered):
//   A tile (activations): global_load_async_to_lds_b128  (ASYNCcnt)
//   B tile (weights)    : TENSOR_LOAD_TO_LDS descriptor  (TENSORcnt, wave 0)
// Math: v_wmma_f32_16x16x32_bf16, fp32 accumulation.
// ---------------------------------------------------------------------------

typedef __attribute__((ext_vector_type(16))) __bf16        v16bf;
typedef __attribute__((ext_vector_type(8)))  float         v8f;
typedef __attribute__((ext_vector_type(4)))  unsigned int  u32x4;
typedef __attribute__((ext_vector_type(4)))  unsigned int  tdm_g0;
typedef __attribute__((ext_vector_type(8)))  int           tdm_g1;
typedef __attribute__((ext_vector_type(4)))  int           tdm_g23;
typedef __attribute__((ext_vector_type(8)))  int           tdm_g4;

#define BM     128
#define BN     128
#define KT     32
#define LDT    40        // LDS row stride in bf16 (32 data + 8 pad; 80 B)

#define M_TOT  16384
#define E_DIM  1024
#define F_DIM  2048

union FragBF  { v16bf v; u32x4 q[2]; };
union Pack8BF { __bf16 h[8]; u32x4 q; };

// --------------------------- address helpers -------------------------------
__device__ __forceinline__ unsigned lds_lo32(const void* p) {
    // AS3 -> generic cast places the LDS byte offset in the low 32 bits.
    return (unsigned)(unsigned long long)p;
}

// ------------------------ async global->LDS (A tiles) ----------------------
__device__ __forceinline__ void async_copy_b128(unsigned lds_addr,
                                                unsigned long long sbase,
                                                unsigned voff) {
    // GVS mode: mem = SADDR + VADDR(u32); 16 B per lane -> LDS[lds_addr].
    asm volatile("global_load_async_to_lds_b128 %0, %1, %2"
                 :: "v"(lds_addr), "v"(voff), "s"(sbase)
                 : "memory");
}
#define S_WAIT_ASYNC_2() asm volatile("s_wait_asynccnt 2" ::: "memory")
#define S_WAIT_ASYNC_0() asm volatile("s_wait_asynccnt 0" ::: "memory")

// A tile = 128 rows x 32 bf16 (4 x 16B chunks per row): 512 lane-copies,
// 256 threads -> 2 async instructions per thread per tile.
template<int KDIM>
__device__ __forceinline__ void stage_A_async(unsigned long long gA, int arow0,
                                              int k0, __bf16* sA, int tid) {
#pragma unroll
    for (int i = 0; i < 2; ++i) {
        const int id = tid + 256 * i;
        const int r  = id >> 2;
        const int c  = (id & 3) * 8;                        // bf16 elems
        async_copy_b128(lds_lo32(sA + r * LDT + c), gA,
                        (unsigned)(((arow0 + r) * KDIM + k0 + c) * 2));
    }
}

// ------------------------ TDM global->LDS (B tiles) ------------------------
// 2D tile (tile_dim0=32 bf16 along K, tile_dim1=128 rows) from a row-major
// [rows][K] bf16 tensor. HW padding: every 16 DWORDs (64 B row) insert
// 4 DWORDs (16 B) -> reproduces the LDT=40 LDS layout.
__device__ __forceinline__ void tdm_load_tile(unsigned lds_addr,
                                              unsigned long long gaddr,
                                              int kdim, int nrows) {
    const tdm_g0 g0 = {
        1u,                                                   // count=1 (user)
        lds_addr,                                             // LDS byte addr
        (unsigned)(gaddr & 0xFFFFFFFFull),                    // global[31:0]
        (unsigned)((gaddr >> 32) & 0x1FFFFFFull) | (2u << 30) // global[56:32], type=2
    };
    const tdm_g1 g1 = {
        (int)((1u << 16) |            // data_size = 2 B
              (1u << 20) |            // pad_enable
              (3u << 22) |            // pad_interval: 16 DWORDs
              (3u << 25)),            // pad_amount  : 4 DWORDs
        (int)(((unsigned)kdim & 0xFFFFu) << 16),                         // tensor_dim0[15:0]
        (int)((((unsigned)kdim >> 16) & 0xFFFFu) |
              (((unsigned)nrows & 0xFFFFu) << 16)),                      // dim0 hi | dim1 lo
        (int)((((unsigned)nrows >> 16) & 0xFFFFu) | ((unsigned)KT << 16)), // dim1 hi | tile_dim0
        (int)(unsigned)BN,                                                // tile_dim1
        kdim,                                                             // dim0 stride lo
        0, 0
    };
    const tdm_g23 z4 = { 0, 0, 0, 0 };
    const tdm_g4  z8 = { 0, 0, 0, 0, 0, 0, 0, 0 };
    __builtin_amdgcn_tensor_load_to_lds(g0, g1, z4, z4, z8, 0);
}

// --------------------------- WMMA fragment loads ---------------------------
// A (16x32 MxK): lane l: half=l>>4, m=l&15; two contiguous 16B runs at
// k = half*8 and k = 16 + half*8.
__device__ __forceinline__ v16bf load_frag_a(const __bf16* sA, int mrow, int lane) {
    const int half = (lane >> 4) & 1;
    const int m    = lane & 15;
    const __bf16* p = sA + (mrow + m) * LDT + half * 8;
    FragBF f;
    f.q[0] = *(const u32x4*)(p);
    f.q[1] = *(const u32x4*)(p + 16);
    return f.v;
}
// B (32x16 KxN), LDS holds weights as [n][k]: lane l: n=l&15, two 16B runs at k = half*8 and k = 16 + half*8.
__device__ __forceinline__ v16bf load_frag_b(const __bf16* sB, int ncol, int lane) {
    const int half = (lane >> 4) & 1;
    const int n    = lane & 15;
    const __bf16* p = sB + (ncol + n) * LDT + half * 8;
    FragBF f;
    f.q[0] = *(const u32x4*)(p);
    f.q[1] = *(const u32x4*)(p + 16);
    return f.v;
}

// Branch-free tanh-GELU: x * sigmoid(1.5957691*(x + 0.044715 x^3))
__device__ __forceinline__ float gelu_fast(float x) {
    float u = x + 0.044715f * x * x * x;
    float e = __expf(-1.595769122f * u);
    return x * __builtin_amdgcn_rcpf(1.0f + e);
}

// --------------------------- prepass kernels -------------------------------
__global__ __launch_bounds__(256)
void k_cvt_bf16(const float* __restrict__ src, __bf16* __restrict__ dst, int n) {
    const int i = (blockIdx.x * 256 + threadIdx.x) * 8;
    if (i >= n) return;
    const float4 a = *(const float4*)(src + i);
    const float4 b = *(const float4*)(src + i + 4);
    Pack8BF p;
    p.h[0] = (__bf16)a.x; p.h[1] = (__bf16)a.y; p.h[2] = (__bf16)a.z; p.h[3] = (__bf16)a.w;
    p.h[4] = (__bf16)b.x; p.h[5] = (__bf16)b.y; p.h[6] = (__bf16)b.z; p.h[7] = (__bf16)b.w;
    *(u32x4*)(dst + i) = p.q;
}

__global__ __launch_bounds__(256)
void k_gather_cvt(const int* __restrict__ idxs, const float* __restrict__ tok,
                  __bf16* __restrict__ xb) {
    const int m = blockIdx.x;                 // one token row per block
    const long src = (long)idxs[m] * F_DIM;
    const long dst = (long)m * F_DIM;
    const int  t   = threadIdx.x * 8;         // 256 thr * 8 = 2048
    const float4 a = *(const float4*)(tok + src + t);
    const float4 b = *(const float4*)(tok + src + t + 4);
    Pack8BF p;
    p.h[0] = (__bf16)a.x; p.h[1] = (__bf16)a.y; p.h[2] = (__bf16)a.z; p.h[3] = (__bf16)a.w;
    p.h[4] = (__bf16)b.x; p.h[5] = (__bf16)b.y; p.h[6] = (__bf16)b.z; p.h[7] = (__bf16)b.w;
    *(u32x4*)(xb + dst + t) = p.q;
}

// ---------------------------------------------------------------------------
// Unified GEMM: out = epilogue(A[M,K]bf16 @ Bw[N,K]bf16^T + bias)
// grid (N/BN, M/BM), 256 threads (8 waves), wave tile 32x64, double buffered.
// ---------------------------------------------------------------------------
template<int KDIM, bool DO_GELU>
__global__ __launch_bounds__(256)
void k_gemm(const __bf16* __restrict__ A, const __bf16* __restrict__ Bw,
            const float* __restrict__ bias,
            __bf16* __restrict__ out_bf, float* __restrict__ out_f32) {
    __shared__ __bf16 sA[2][BM * LDT];
    __shared__ __bf16 sB[2][BN * LDT];

    const int tid  = threadIdx.x;
    const int lane = tid & 31;
    const int wave = tid >> 5;
    const int wm   = wave & 3;
    const int wn   = wave >> 2;
    const int mblk = blockIdx.y * BM;
    const int nblk = blockIdx.x * BN;
    const unsigned long long gA = (unsigned long long)A;
    const unsigned long long gB = (unsigned long long)Bw;

    v8f c[2][4] = {};

    constexpr int NT = KDIM / KT;
    stage_A_async<KDIM>(gA, mblk, 0, sA[0], tid);
    if (wave == 0)
        tdm_load_tile(lds_lo32(sB[0]),
                      gB + (unsigned long long)((long)nblk * KDIM) * 2,
                      KDIM, E_DIM);

    for (int it = 0; it < NT; ++it) {
        const int buf = it & 1;
        if (it + 1 < NT) {
            stage_A_async<KDIM>(gA, mblk, (it + 1) * KT, sA[buf ^ 1], tid);
            if (wave == 0)
                tdm_load_tile(lds_lo32(sB[buf ^ 1]),
                              gB + (unsigned long long)((long)nblk * KDIM +
                                                        (it + 1) * KT) * 2,
                              KDIM, E_DIM);
            S_WAIT_ASYNC_2();                                   // tile `it` A done
            if (wave == 0) __builtin_amdgcn_s_wait_tensorcnt(1); // tile `it` B done
        } else {
            S_WAIT_ASYNC_0();
            if (wave == 0) __builtin_amdgcn_s_wait_tensorcnt(0);
        }
        __syncthreads();

        v16bf af[2];
#pragma unroll
        for (int tm = 0; tm < 2; ++tm)
            af[tm] = load_frag_a(sA[buf], wm * 32 + tm * 16, lane);
        v16bf bfrag[4];
#pragma unroll
        for (int tn = 0; tn < 4; ++tn)
            bfrag[tn] = load_frag_b(sB[buf], wn * 64 + tn * 16, lane);

#pragma unroll
        for (int tm = 0; tm < 2; ++tm)
#pragma unroll
            for (int tn = 0; tn < 4; ++tn)
                c[tm][tn] = __builtin_amdgcn_wmma_f32_16x16x32_bf16(
                    false, af[tm], false, bfrag[tn], (short)0, c[tm][tn],
                    false, false);
        __syncthreads();
    }

    const int half = (lane >> 4) & 1;
    const int nl   = lane & 15;
#pragma unroll
    for (int tm = 0; tm < 2; ++tm)
#pragma unroll
        for (int tn = 0; tn < 4; ++tn) {
            const int ng = nblk + wn * 64 + tn * 16 + nl;
            const int m0 = mblk + wm * 32 + tm * 16 + half * 8;
            const float b = bias[ng];
#pragma unroll
            for (int v = 0; v < 8; ++v) {
                const float x = c[tm][tn][v] + b;
                if constexpr (DO_GELU)
                    out_bf[(m0 + v) * E_DIM + ng] = (__bf16)gelu_fast(x);
                else
                    out_f32[(m0 + v) * E_DIM + ng] = x;
            }
        }
}

// ---------------------------------------------------------------------------
extern "C" void kernel_launch(void* const* d_in, const int* in_sizes, int n_in,
                              void* d_out, int out_size, void* d_ws, size_t ws_size,
                              hipStream_t stream) {
    (void)in_sizes; (void)n_in; (void)out_size; (void)ws_size;

    const int*   idxs      = (const int*)  d_in[0];
    const float* tok_embed = (const float*)d_in[1];
    const float* W1        = (const float*)d_in[2];
    const float* b1        = (const float*)d_in[3];
    const float* W2        = (const float*)d_in[4];
    const float* b2        = (const float*)d_in[5];
    float*       out       = (float*)d_out;

    // workspace layout (bytes):
    //   [0, 32M)        act  bf16 [16384][1024]
    //   [32M, 96M)      xb   bf16 [16384][2048]
    //   [96M, 100M)     w1b  bf16 [1024][2048]
    //   [100M, 102M)    w2b  bf16 [1024][1024]
    char* ws = (char*)d_ws;
    __bf16* act = (__bf16*)(ws);
    __bf16* xb  = (__bf16*)(ws + (size_t)33554432);
    __bf16* w1b = (__bf16*)(ws + (size_t)33554432 + 67108864);
    __bf16* w2b = (__bf16*)(ws + (size_t)33554432 + 67108864 + 4194304);

    // prepass: weight conversion + gathered-embedding conversion
    k_cvt_bf16<<<dim3((E_DIM * F_DIM) / (256 * 8)), dim3(256), 0, stream>>>(W1, w1b, E_DIM * F_DIM);
    k_cvt_bf16<<<dim3((E_DIM * E_DIM) / (256 * 8)), dim3(256), 0, stream>>>(W2, w2b, E_DIM * E_DIM);
    k_gather_cvt<<<dim3(M_TOT), dim3(256), 0, stream>>>(idxs, tok_embed, xb);

    dim3 block(256);
    dim3 grid(E_DIM / BN, M_TOT / BM);        // (8, 128)
    k_gemm<F_DIM, true ><<<grid, block, 0, stream>>>(xb,  w1b, b1, act, nullptr);
    k_gemm<E_DIM, false><<<grid, block, 0, stream>>>(act, w2b, b2, nullptr, out);
}
